// FullHyperPQhead_73108933312727
// MI455X (gfx1250) — matrix operand order repaired and checked by reading, not compiled, not run on hardware
//
#include <hip/hip_runtime.h>
#include <hip/hip_bf16.h>
#include <math.h>

// ---------------- problem constants (match reference) ----------------
#define PM      16          // heads
#define PD      64          // dim per head
#define PFEAT   (PM * PD)   // 1024
#define PK      1024        // codebook size
#define PN      8192        // batch
#define PTEMP   10.0f
#define PCLIP   1.0f
#define PEPS    1e-8f

// output layout (floats) in d_out, tuple order: hx, xhat, codes, soft, qerr
#define OFF_HX    ((size_t)0)
#define OFF_XHAT  ((size_t)PN * PM * PD)                  // 8388608
#define OFF_CODES ((size_t)2 * PN * PM * PD)              // 16777216
#define OFF_SOFT  (OFF_CODES + (size_t)PN * PM)           // 16908288
#define OFF_QERR  (OFF_SOFT + (size_t)PN * PM * PK)       // 151126016

// workspace layout (bytes)
#define WS_HC    ((size_t)0)                               // K*M*D f16 = 2 MB  [k][m][d]
#define WS_HCT   ((size_t)2 << 20)                         // 2 MB              [m][d][k]
#define WS_HXG   ((size_t)4 << 20)                         // N*M*D f16 = 16 MB [n][m][d] (g applied)
#define WS_QPART ((size_t)20 << 20)                        // 8192 f32 partials

typedef __attribute__((ext_vector_type(16))) _Float16 v16h;
typedef __attribute__((ext_vector_type(8)))  _Float16 v8h;
typedef __attribute__((ext_vector_type(4)))  _Float16 v4h;
typedef __attribute__((ext_vector_type(8)))  float    v8f;
typedef __attribute__((ext_vector_type(4)))  float    v4f;

union Frag16 { v16h v; v8h h[2]; };

__device__ __forceinline__ float wave_reduce_add(float v) {
  #pragma unroll
  for (int off = 16; off > 0; off >>= 1) v += __shfl_xor(v, off, 32);
  return v;
}

// ---------------------------------------------------------------------
// expmap0 prep for codebook: writes hc_f16 [k][m][d] and hcT_f16 [m][d][k]
// one wave32 per (k,m) vector; lane handles d = lane and d = lane+32
// ---------------------------------------------------------------------
__global__ __launch_bounds__(128) void prep_hc_kernel(
    const float* __restrict__ C, const float* __restrict__ neg_curvs,
    _Float16* __restrict__ hc, _Float16* __restrict__ hcT) {
  const int vec  = blockIdx.x * 4 + (threadIdx.x >> 5);
  const int lane = threadIdx.x & 31;
  const int k = vec >> 4, m = vec & 15;
  const int d0 = lane, d1 = lane + 32;

  float v0 = (d0 == 0) ? 0.0f : C[(size_t)k * PFEAT + m * PD + d0];
  float v1 = C[(size_t)k * PFEAT + m * PD + d1];
  float ss = wave_reduce_add(v0 * v0 + v1 * v1);
  float vn = sqrtf(ss);
  float sc = fminf(1.0f, PCLIP / fmaxf(vn, PEPS));
  v0 *= sc; v1 *= sc;
  float xn = fmaxf(vn * sc, PEPS);

  float c  = neg_curvs[m];
  float sK = sqrtf(1.0f / c);
  float th = xn / sK;
  float coef = sK * sinhf(th) / xn;
  float o0 = (d0 == 0) ? sK * coshf(th) : coef * v0;
  float o1 = coef * v1;

  hc[((size_t)k * PM + m) * PD + d0] = (_Float16)o0;
  hc[((size_t)k * PM + m) * PD + d1] = (_Float16)o1;
  hcT[((size_t)m * PD + d0) * PK + k] = (_Float16)o0;
  hcT[((size_t)m * PD + d1) * PK + k] = (_Float16)o1;
}

// ---------------------------------------------------------------------
// expmap0 prep for x: writes hx fp32 (output 0) and hxg_f16 = hx * g
// ---------------------------------------------------------------------
__global__ __launch_bounds__(128) void prep_hx_kernel(
    const float* __restrict__ x, const float* __restrict__ neg_curvs,
    float* __restrict__ out_hx, _Float16* __restrict__ hxg) {
  const int vec  = blockIdx.x * 4 + (threadIdx.x >> 5);
  const int lane = threadIdx.x & 31;
  const int n = vec >> 4, m = vec & 15;
  const int d0 = lane, d1 = lane + 32;

  float v0 = (d0 == 0) ? 0.0f : x[(size_t)n * PFEAT + m * PD + d0];
  float v1 = x[(size_t)n * PFEAT + m * PD + d1];
  float ss = wave_reduce_add(v0 * v0 + v1 * v1);
  float vn = sqrtf(ss);
  float sc = fminf(1.0f, PCLIP / fmaxf(vn, PEPS));
  v0 *= sc; v1 *= sc;
  float xn = fmaxf(vn * sc, PEPS);

  float c  = neg_curvs[m];
  float sK = sqrtf(1.0f / c);
  float th = xn / sK;
  float coef = sK * sinhf(th) / xn;
  float o0 = (d0 == 0) ? sK * coshf(th) : coef * v0;
  float o1 = coef * v1;

  size_t base = ((size_t)n * PM + m) * PD;
  out_hx[base + d0] = o0;
  out_hx[base + d1] = o1;
  hxg[base + d0] = (_Float16)((d0 == 0) ? -o0 : o0);  // Minkowski g applied
  hxg[base + d1] = (_Float16)o1;
}

// ---------------------------------------------------------------------
// fused main kernel: one block = (16-row n-tile, head m), 4 waves
//   GEMM1 (WMMA f16, 2 tiles in flight): logits 16x1024 into LDS (f32)
//   softmax + argmax (float4 LDS), soft -> global f32 + LDS f16
//   GEMM2 (WMMA f16, A from f16 LDS via b128): mid -> xhat, qerr partial
// ---------------------------------------------------------------------
#define LSTR 1028   // f32 logits row stride: 8*1028 % 64 == 32 (no half-wave conflict)
#define SSTR 1032   // f16 soft row stride (halves): 4-bank rotation per row
#define MSTR 68

__global__ __launch_bounds__(128) void hyperpq_main_kernel(
    const _Float16* __restrict__ hxg, const _Float16* __restrict__ hc,
    const _Float16* __restrict__ hcT, const float* __restrict__ neg_curvs,
    const float* __restrict__ hx_f32,
    float* __restrict__ out_xhat, int* __restrict__ out_codes,
    float* __restrict__ out_soft, float* __restrict__ qpart) {

  __shared__ float     lds_l[16 * LSTR];    // logits -> exp(f32)   (64.25 KB)
  __shared__ _Float16  lds_s16[16 * SSTR];  // soft f16 for GEMM2 A (32.25 KB)
  __shared__ float     lds_mid[16 * MSTR];  // mid 16x64 (padded)
  __shared__ float red_max[16][8];
  __shared__ int   red_idx[16][8];
  __shared__ float red_sum[16][8];
  __shared__ float red_mi[16][8];
  __shared__ float red_q[16][8];
  __shared__ float row_a[16];               // rowmax, then 1/rowsum
  __shared__ float row_scale[16];
  __shared__ float qacc[16];

  const int t    = threadIdx.x;
  const int lane = t & 31;
  const int wave = t >> 5;
  const int hh   = lane >> 4;     // K-half selector
  const int col  = lane & 15;     // tile column / row id

  const int n0 = blockIdx.x * 16;
  const int m  = blockIdx.y;
  const float cM  = neg_curvs[m];
  const float Kc  = 1.0f / cM;
  const float sK  = sqrtf(Kc);

  // ---------------- GEMM1: inn = hxg(16x64) @ hc_m^T(64x1024) ----------------
  // A fragments (16x32 f16): row = col, K = 16h..16h+15 contiguous
  const _Float16* pa = hxg + ((size_t)(n0 + col) * PM + m) * PD;
  Frag16 a0, a1;
  a0.h[0] = *(const v8h*)(pa + 16 * hh);
  a0.h[1] = *(const v8h*)(pa + 16 * hh + 8);
  a1.h[0] = *(const v8h*)(pa + 32 + 16 * hh);
  a1.h[1] = *(const v8h*)(pa + 32 + 16 * hh + 8);

  // two k-tiles per iteration: independent accumulators hide WMMA->VALU hazard
  for (int kt0 = wave; kt0 < PK / 16; kt0 += 8) {
    const int ktA = kt0, ktB = kt0 + 4;
    const _Float16* pbA = hc + ((size_t)(ktA * 16 + col) * PM + m) * PD;
    const _Float16* pbB = hc + ((size_t)(ktB * 16 + col) * PM + m) * PD;
    __builtin_prefetch(pbA + (size_t)128 * PM * PD, 0, 1);  // next iter's lines
    Frag16 bA0, bA1, bB0, bB1;
    bA0.h[0] = *(const v8h*)(pbA + 16 * hh);
    bA0.h[1] = *(const v8h*)(pbA + 16 * hh + 8);
    bA1.h[0] = *(const v8h*)(pbA + 32 + 16 * hh);
    bA1.h[1] = *(const v8h*)(pbA + 32 + 16 * hh + 8);
    bB0.h[0] = *(const v8h*)(pbB + 16 * hh);
    bB0.h[1] = *(const v8h*)(pbB + 16 * hh + 8);
    bB1.h[0] = *(const v8h*)(pbB + 32 + 16 * hh);
    bB1.h[1] = *(const v8h*)(pbB + 32 + 16 * hh + 8);

    v8f accA = {}, accB = {};
    accA = __builtin_amdgcn_wmma_f32_16x16x32_f16(false, a0.v, false, bA0.v,
                                                  (short)0, accA, false, false);
    accB = __builtin_amdgcn_wmma_f32_16x16x32_f16(false, a0.v, false, bB0.v,
                                                  (short)0, accB, false, false);
    accA = __builtin_amdgcn_wmma_f32_16x16x32_f16(false, a1.v, false, bA1.v,
                                                  (short)0, accA, false, false);
    accB = __builtin_amdgcn_wmma_f32_16x16x32_f16(false, a1.v, false, bB1.v,
                                                  (short)0, accB, false, false);
    #pragma unroll
    for (int j = 0; j < 8; ++j) {               // D row M = j + 8h, col N = col
      float sqA = fmaxf(-2.0f * Kc - 2.0f * accA[j], PEPS);
      float sqB = fmaxf(-2.0f * Kc - 2.0f * accB[j], PEPS);
      lds_l[(j + 8 * hh) * LSTR + ktA * 16 + col] = -__builtin_amdgcn_sqrtf(sqA);
      lds_l[(j + 8 * hh) * LSTR + ktB * 16 + col] = -__builtin_amdgcn_sqrtf(sqB);
    }
  }
  __syncthreads();

  // ---------------- softmax over K=1024 per row (+ argmax) ----------------
  const int r  = t >> 3;        // row 0..15
  const int gi = t & 7;         // 8 threads per row, 128-wide chunks
  const int kb = gi * 128;
  {
    float bm = -1e30f; int bi = 0;
    for (int k4 = 0; k4 < 32; ++k4) {
      v4f v = *(const v4f*)&lds_l[r * LSTR + kb + 4 * k4];
      #pragma unroll
      for (int e = 0; e < 4; ++e)
        if (v[e] > bm) { bm = v[e]; bi = kb + 4 * k4 + e; }
    }
    red_max[r][gi] = bm; red_idx[r][gi] = bi;
  }
  __syncthreads();
  if (t < 16) {
    float bm = red_max[t][0]; int bi = red_idx[t][0];
    #pragma unroll
    for (int g2 = 1; g2 < 8; ++g2)
      if (red_max[t][g2] > bm) { bm = red_max[t][g2]; bi = red_idx[t][g2]; }
    row_a[t] = bm;
    out_codes[(size_t)(n0 + t) * PM + m] = bi;
  }
  __syncthreads();
  {
    float rm = row_a[r], s = 0.0f;
    for (int k4 = 0; k4 < 32; ++k4) {
      float* p = &lds_l[r * LSTR + kb + 4 * k4];
      v4f v = *(const v4f*)p;
      #pragma unroll
      for (int e = 0; e < 4; ++e) { v[e] = __expf(PTEMP * (v[e] - rm)); s += v[e]; }
      *(v4f*)p = v;
    }
    red_sum[r][gi] = s;
  }
  __syncthreads();
  if (t < 16) {
    float s = 0.0f;
    #pragma unroll
    for (int g2 = 0; g2 < 8; ++g2) s += red_sum[t][g2];
    row_a[t] = __builtin_amdgcn_rcpf(s);
  }
  __syncthreads();
  // normalize: float4 store to global soft + f16 copy into LDS for GEMM2 A
  for (int i = 0; i < 32; ++i) {
    int f4 = t + 128 * i;               // 4096 float4s = 16x1024
    int rr = f4 >> 8, k = (f4 & 255) * 4;
    v4f v = *(const v4f*)&lds_l[rr * LSTR + k];
    float ri = row_a[rr];
    v4h hv;
    #pragma unroll
    for (int e = 0; e < 4; ++e) { v[e] *= ri; hv[e] = (_Float16)v[e]; }
    *(v4f*)(out_soft + (size_t)(n0 + rr) * (PM * PK) + (size_t)m * PK + k) = v;
    *(v4h*)&lds_s16[rr * SSTR + k] = hv;
  }
  __syncthreads();

  // ---------------- GEMM2: mid = soft(16x1024) @ hc_m(1024x64) ----------------
  {
    v8f acc = {};
    const _Float16* ps = lds_s16 + (size_t)col * SSTR;   // A row = lane&15
    const _Float16* pbbase =
        hcT + ((size_t)m * PD + wave * 16 + col) * PK + 16 * hh;
    for (int step = 0; step < PK / 32; ++step) {
      const int kk0 = step * 32;
      Frag16 a, b;
      a.h[0] = *(const v8h*)(ps + kk0 + 16 * hh);
      a.h[1] = *(const v8h*)(ps + kk0 + 16 * hh + 8);
      b.h[0] = *(const v8h*)(pbbase + kk0);
      b.h[1] = *(const v8h*)(pbbase + kk0 + 8);
      acc = __builtin_amdgcn_wmma_f32_16x16x32_f16(false, a.v, false, b.v,
                                                   (short)0, acc, false, false);
    }
    #pragma unroll
    for (int j = 0; j < 8; ++j)
      lds_mid[(j + 8 * hh) * MSTR + wave * 16 + col] = acc[j];
  }
  __syncthreads();

  // ---------------- epilogue: xhat, qerr partial ----------------
  {
    float mi = 0.0f, qp = 0.0f;
    const float* phx = hx_f32 + ((size_t)(n0 + r) * PM + m) * PD + gi * 8;
    #pragma unroll
    for (int j = 0; j < 8; ++j) {
      int d = gi * 8 + j;
      float g  = (d == 0) ? -1.0f : 1.0f;
      float md = lds_mid[r * MSTR + d];
      mi += g * md * md;
      qp += g * phx[j] * md;
    }
    red_mi[r][gi] = mi; red_q[r][gi] = qp;
  }
  __syncthreads();
  if (t < 16) {
    float smi = 0.0f, sq = 0.0f;
    #pragma unroll
    for (int g2 = 0; g2 < 8; ++g2) { smi += red_mi[t][g2]; sq += red_q[t][g2]; }
    float scl = sK * __builtin_amdgcn_rsqf(fmaxf(fabsf(smi), PEPS));
    row_scale[t] = scl;
    qacc[t] = -2.0f * Kc - 2.0f * (scl * sq);
  }
  __syncthreads();
  #pragma unroll
  for (int j = 0; j < 8; ++j) {                    // 16x64 xhat values
    int idx = t * 8 + j;
    int rr = idx >> 6, d = idx & 63;
    out_xhat[((size_t)(n0 + rr) * PM + m) * PD + d] =
        row_scale[rr] * lds_mid[rr * MSTR + d];
  }
  if (t == 0) {
    float s = 0.0f;
    #pragma unroll
    for (int i = 0; i < 16; ++i) s += qacc[i];
    qpart[blockIdx.y * gridDim.x + blockIdx.x] = s;
  }
}

// deterministic qerr reduction over 8192 block partials
__global__ __launch_bounds__(256) void qerr_reduce_kernel(
    const float* __restrict__ qpart, float* __restrict__ out_qerr) {
  __shared__ float sh[256];
  float s = 0.0f;
  for (int i = threadIdx.x; i < 8192; i += 256) s += qpart[i];
  sh[threadIdx.x] = s;
  __syncthreads();
  for (int off = 128; off > 0; off >>= 1) {
    if (threadIdx.x < off) sh[threadIdx.x] += sh[threadIdx.x + off];
    __syncthreads();
  }
  if (threadIdx.x == 0) *out_qerr = sh[0] / (float)((size_t)PN * PM);
}

// ---------------------------------------------------------------------
extern "C" void kernel_launch(void* const* d_in, const int* in_sizes, int n_in,
                              void* d_out, int out_size, void* d_ws, size_t ws_size,
                              hipStream_t stream) {
  (void)in_sizes; (void)n_in; (void)out_size; (void)ws_size;
  const float* x   = (const float*)d_in[0];
  const float* C   = (const float*)d_in[1];
  const float* ncv = (const float*)d_in[2];
  float* out = (float*)d_out;

  _Float16* hc    = (_Float16*)((char*)d_ws + WS_HC);
  _Float16* hcT   = (_Float16*)((char*)d_ws + WS_HCT);
  _Float16* hxg   = (_Float16*)((char*)d_ws + WS_HXG);
  float*    qpart = (float*)((char*)d_ws + WS_QPART);

  // 1) codebook expmap -> f16 (both layouts)
  prep_hc_kernel<<<(PK * PM) / 4, 128, 0, stream>>>(C, ncv, hc, hcT);
  // 2) x expmap -> hx fp32 (output) + hxg f16
  prep_hx_kernel<<<(PN * PM) / 4, 128, 0, stream>>>(x, ncv, out + OFF_HX, hxg);
  // 3) fused WMMA pipeline
  dim3 grid(PN / 16, PM);
  hyperpq_main_kernel<<<grid, 128, 0, stream>>>(
      hxg, hc, hcT, ncv, out + OFF_HX,
      out + OFF_XHAT, (int*)(out + OFF_CODES), out + OFF_SOFT, qpart);
  // 4) deterministic qerr
  qerr_reduce_kernel<<<1, 256, 0, stream>>>(qpart, out + OFF_QERR);
}
